// TriangularAttentionStartingNode_18528488915231
// MI455X (gfx1250) — matrix operand order repaired and checked by reading, not compiled, not run on hardware
//
#include <hip/hip_runtime.h>
#include <hip/hip_fp16.h>

#define R  256
#define RR 65536
#define CZ 128
#define C  32
#define H  4

typedef __attribute__((ext_vector_type(16))) _Float16 v16h;
typedef __attribute__((ext_vector_type(8)))  _Float16 v8h;
typedef __attribute__((ext_vector_type(8)))  float    v8f;
typedef __attribute__((ext_vector_type(4)))  unsigned int v4u;
typedef __attribute__((ext_vector_type(8)))  int v8i;
typedef __attribute__((ext_vector_type(4)))  int v4i;

#if __has_builtin(__builtin_amdgcn_tensor_load_to_lds)
#define HAS_TDM 1
#else
#define HAS_TDM 0
#endif

static __device__ __forceinline__ v8f wmma_f32_f16(v16h a, v16h b, v8f c) {
    // D = A(16x32 f16) * B(32x16 f16) + C(16x16 f32)
    return __builtin_amdgcn_wmma_f32_16x16x32_f16(false, a, false, b, (short)0, c, false, false);
}

// A fragment: 16(M) x 32(K), src row-major [M][K], leading dim ld (halves).
// m=lane&15; half t -> K = kb + t, kb=(lane>>4)*16 (contiguous, matches B fragments).
static __device__ __forceinline__ v16h load_a_frag(const _Float16* src, int ld) {
    const int lane = threadIdx.x & 31;
    const int m  = lane & 15;
    const int kb = (lane >> 4) * 16;
    const _Float16* p = src + (size_t)m * ld + kb;
    v8h lo = *(const v8h*)p;
    v8h hi = *(const v8h*)(p + 8);
    v16h f;
#pragma unroll
    for (int t = 0; t < 8; ++t) { f[t] = lo[t]; f[t + 8] = hi[t]; }
    return f;
}

// B fragment: 32(K) x 16(N); src stored as [N][K] row-major (src[n*ld+k] = B[k][n]).
// n=lane&15; half t -> K = (lane>>4)*16 + t  => one contiguous 32B load per lane.
static __device__ __forceinline__ v16h load_b_frag(const _Float16* src, int ld) {
    const int lane = threadIdx.x & 31;
    const int n  = lane & 15;
    const int kb = (lane >> 4) * 16;
    return *(const v16h*)(src + (size_t)n * ld + kb);
}

#if HAS_TDM
// Issue a TDM 2-D tile load (f16 elements): tile_w x tile_h, row stride in halves.
// D# per CDNA5 ISA 8.3/8.4: group0 {count=1, lds_addr, global_addr, type=2},
// group1 {mask=0, data_size=1(2B), dims, dim0_stride}; groups 2/3 zero (<=2D).
static __device__ __forceinline__ void tdm_issue(const _Float16* gsrc, unsigned lds_off,
                                                 unsigned tile_w, unsigned tile_h,
                                                 unsigned long long row_stride_halves) {
    const unsigned long long ga = (unsigned long long)(uintptr_t)gsrc;
    v4u g0;
    g0.x = 1u;                                                  // count=1, user mode
    g0.y = lds_off;                                             // lds_addr (bytes)
    g0.z = (unsigned)ga;                                        // global_addr[31:0]
    g0.w = (unsigned)((ga >> 32) & 0x01FFFFFFu) | (2u << 30);   // addr[56:32] | type=2
    v8i g1;
    g1[0] = (int)(1u << 16);                                    // data_size=1 (2B), mask=0
    g1[1] = (int)((tile_w & 0xFFFFu) << 16);                    // tensor_dim0[15:0]
    g1[2] = (int)((tile_w >> 16) | ((tile_h & 0xFFFFu) << 16)); // dim0[31:16] | tensor_dim1[15:0]
    g1[3] = (int)(((tile_h >> 16) & 0xFFFFu) | ((tile_w & 0xFFFFu) << 16)); // dim1 hi | tile_dim0
    g1[4] = (int)(tile_h & 0xFFFFu);                            // tile_dim1 (tile_dim2=0)
    g1[5] = (int)(unsigned)(row_stride_halves & 0xFFFFFFFFull); // tensor_dim0_stride lo
    g1[6] = (int)(unsigned)((row_stride_halves >> 32) & 0xFFFFull);
    g1[7] = 0;
    v4i z4 = {0, 0, 0, 0};
#if __clang_major__ >= 23
    v8i z8 = {0, 0, 0, 0, 0, 0, 0, 0};
    __builtin_amdgcn_tensor_load_to_lds(g0, g1, z4, z4, z8, 0);
#else
    __builtin_amdgcn_tensor_load_to_lds(g0, g1, z4, z4, 0);
#endif
}
#endif

static __device__ __forceinline__ float block_reduce_sum_128(float v, float* sm) {
#pragma unroll
    for (int m = 16; m >= 1; m >>= 1) v += __shfl_xor(v, m, 32);
    if ((threadIdx.x & 31) == 0) sm[threadIdx.x >> 5] = v;
    __syncthreads();
    float t = sm[0] + sm[1] + sm[2] + sm[3];
    __syncthreads();
    return t;
}

// ---------------------------------------------------------------------------
// Kernel 1: fold LayerNorm affine into weights, convert to f16.
//   W'[h,c,z] = W[h,c,z]*gamma[h,z]  (f16),  b'[h,c] = b[h,c] + sum_z beta[h,z]*W[h,c,z]
// grid.x = 512 (q,k,v,g rows) + 1024 (Wb rows) + 128 (Wout convert), block 128
// ---------------------------------------------------------------------------
__global__ void ta_prep_kernel(
    const float* __restrict__ gamma, const float* __restrict__ beta,
    const float* __restrict__ Wq, const float* __restrict__ bq,
    const float* __restrict__ Wk, const float* __restrict__ bk,
    const float* __restrict__ Wv, const float* __restrict__ bv,
    const float* __restrict__ Wb,
    const float* __restrict__ Wg, const float* __restrict__ bg,
    const float* __restrict__ Wout,
    _Float16* __restrict__ Wq16, _Float16* __restrict__ Wk16,
    _Float16* __restrict__ Wv16, _Float16* __restrict__ Wg16,
    _Float16* __restrict__ Wb16, _Float16* __restrict__ Wout16,
    float* __restrict__ bqf, float* __restrict__ bkf,
    float* __restrict__ bvf, float* __restrict__ bgf, float* __restrict__ bbf)
{
    __shared__ float sm[4];
    const int b = blockIdx.x;
    const int z = threadIdx.x;
    if (b < 512) {
        const int which = b >> 7;
        const int row = b & 127;                 // h*C + c
        const int h = row >> 5;
        const float* W; const float* bs; _Float16* W16; float* bf;
        if (which == 0)      { W = Wq; bs = bq; W16 = Wq16; bf = bqf; }
        else if (which == 1) { W = Wk; bs = bk; W16 = Wk16; bf = bkf; }
        else if (which == 2) { W = Wv; bs = bv; W16 = Wv16; bf = bvf; }
        else                 { W = Wg; bs = bg; W16 = Wg16; bf = bgf; }
        const float w = W[(size_t)row * CZ + z];
        W16[(size_t)row * CZ + z] = (_Float16)(w * gamma[h * CZ + z]);
        const float tot = block_reduce_sum_128(w * beta[h * CZ + z], sm);
        if (z == 0) bf[row] = bs[row] + tot;
    } else if (b < 1536) {
        const int row = b - 512;                 // h*R + r
        const int h = row >> 8;
        const float w = Wb[(size_t)row * CZ + z];
        Wb16[(size_t)row * CZ + z] = (_Float16)(w * gamma[h * CZ + z]);
        const float tot = block_reduce_sum_128(w * beta[h * CZ + z], sm);
        if (z == 0) bbf[row] = tot;              // Wb has no base bias
    } else {
        const int row = b - 1536;                // Wout [CZ][H*C] -> f16, same layout
        Wout16[(size_t)row * CZ + z] = (_Float16)Wout[(size_t)row * CZ + z];
    }
}

// ---------------------------------------------------------------------------
// Kernel 2: shared LayerNorm -> xn16 [RR, CZ] f16. grid RR, block 128.
// ---------------------------------------------------------------------------
__global__ void ta_ln_kernel(const float* __restrict__ pair, _Float16* __restrict__ xn)
{
    __shared__ float sm[4];
    const size_t row = blockIdx.x;
    const int z = threadIdx.x;
    const float x = pair[row * CZ + z];
    const float mean = block_reduce_sum_128(x, sm) * (1.0f / CZ);
    const float d = x - mean;
    const float var = block_reduce_sum_128(d * d, sm) * (1.0f / CZ);
    xn[row * CZ + z] = (_Float16)(d * rsqrtf(var + 1e-5f));
}

// ---------------------------------------------------------------------------
// Kernel 3: projections. q,k,g computed TRANSPOSED (D[c][row] = W'*xn^T) so
// each lane's 8 C-values are contiguous channels of one row -> pure b128
// stores into row-major [H][RR][C]. v computed normally -> contiguous b128
// stores into transposed [H][C][RR]. xn B-fragments shared across q,k,g.
// grid (RR/64, H), block 128 (4 waves x 16 rows).
// ---------------------------------------------------------------------------
__global__ void ta_proj_kernel(
    const _Float16* __restrict__ xn,
    const _Float16* __restrict__ Wq16, const _Float16* __restrict__ Wk16,
    const _Float16* __restrict__ Wv16, const _Float16* __restrict__ Wg16,
    const float* __restrict__ bqf, const float* __restrict__ bkf,
    const float* __restrict__ bvf, const float* __restrict__ bgf,
    _Float16* __restrict__ q16, _Float16* __restrict__ k16,
    _Float16* __restrict__ vT16, _Float16* __restrict__ g16)
{
    const int wave = threadIdx.x >> 5;
    const int lane = threadIdx.x & 31;
    const int n  = lane & 15;
    const int mb = (lane >> 4) * 8;
    const int rowBase = blockIdx.x * 64 + wave * 16;
    const int h = blockIdx.y;

    // xn as B fragments (shared by q,k,g) and as A fragments (for v)
    v16h bx[4], ax[4];
#pragma unroll
    for (int kc = 0; kc < 4; ++kc) {
        bx[kc] = load_b_frag(xn + (size_t)rowBase * CZ + kc * 32, CZ);
        ax[kc] = load_a_frag(xn + (size_t)rowBase * CZ + kc * 32, CZ);
    }

    // q, k, g : D[c][row]
#pragma unroll
    for (int proj = 0; proj < 3; ++proj) {
        const _Float16* W16 = (proj == 0) ? Wq16 : (proj == 1) ? Wk16 : Wg16;
        const float* bf     = (proj == 0) ? bqf  : (proj == 1) ? bkf  : bgf;
        _Float16* dst       = (proj == 0) ? q16  : (proj == 1) ? k16  : g16;
#pragma unroll
        for (int cb = 0; cb < 2; ++cb) {
            v8f acc = *(const v8f*)(bf + h * C + cb * 16 + mb);   // bias per c = M
#pragma unroll
            for (int kc = 0; kc < 4; ++kc) {
                v16h aw = load_a_frag(W16 + (size_t)(h * C + cb * 16) * CZ + kc * 32, CZ);
                acc = wmma_f32_f16(aw, bx[kc], acc);
            }
            v8h ov;
            if (proj == 2) {
#pragma unroll
                for (int r = 0; r < 8; ++r) ov[r] = (_Float16)(1.0f / (1.0f + __expf(-acc[r])));
            } else {
#pragma unroll
                for (int r = 0; r < 8; ++r) ov[r] = (_Float16)acc[r];
            }
            *(v8h*)(dst + ((size_t)h * RR + rowBase + n) * C + cb * 16 + mb) = ov;
        }
    }

    // v : D[row][c] -> transposed store vT[h][c][row]
#pragma unroll
    for (int nt = 0; nt < 2; ++nt) {
        v8f acc;
        const float bz = bvf[h * C + nt * 16 + n];
#pragma unroll
        for (int r = 0; r < 8; ++r) acc[r] = bz;
#pragma unroll
        for (int kc = 0; kc < 4; ++kc) {
            v16h bw = load_b_frag(Wv16 + (size_t)(h * C + nt * 16) * CZ + kc * 32, CZ);
            acc = wmma_f32_f16(ax[kc], bw, acc);
        }
        v8h ov;
#pragma unroll
        for (int r = 0; r < 8; ++r) ov[r] = (_Float16)acc[r];
        *(v8h*)(vT16 + (size_t)(h * C + nt * 16 + n) * RR + rowBase + mb) = ov;
    }
}

// ---------------------------------------------------------------------------
// Kernel 4: fused pair-bias GEMM + flash attention + gate.
// S^T[y,x] = scale*K·Q^T + Wb'·xn^T + bb'. The wave-shared K/V^T/Wb' slab
// tiles (12KB per 32-y slab) are staged in LDS once per workgroup — via the
// Tensor Data Mover (double-buffered, TENSORcnt-synchronized) when available,
// else a cooperative vector-load fill — eliminating the 8x per-wave redundant
// global traffic. grid (R, H), block 256 (8 waves x 32 x's).
// ---------------------------------------------------------------------------
__global__ void ta_attn_kernel(
    const _Float16* __restrict__ xn,
    const _Float16* __restrict__ q16, const _Float16* __restrict__ k16,
    const _Float16* __restrict__ vT16, const _Float16* __restrict__ g16,
    const _Float16* __restrict__ Wb16, const float* __restrict__ bbf,
    _Float16* __restrict__ ocat)
{
    __shared__ _Float16 smK[2][32 * 32];    // [y 32][c 32]
    __shared__ _Float16 smV[2][32 * 32];    // [c 32][y 32]
    __shared__ _Float16 smWb[2][32 * 128];  // [y 32][z 128]

    const int i = blockIdx.x;
    const int h = blockIdx.y;
    const int wave = threadIdx.x >> 5;
    const int lane = threadIdx.x & 31;
    const int n  = lane & 15;
    const int hi = lane >> 4;
    const int mb = hi * 8;
    const int xBase = wave * 32;
    const size_t rowQ = (size_t)h * RR + (size_t)i * R;   // q/k/g row base (i*R + y)

    // Q fragments (B layout: n->x, K->c), pre-scaled by 1/sqrt(C);
    // xn fragments (B layout: n->x, K->z), rows (x*R + i) for the pair bias.
    v16h qf[2], xnf[2][4];
#pragma unroll
    for (int xs = 0; xs < 2; ++xs) {
        const int xb = xBase + xs * 16;
        qf[xs] = *(const v16h*)(q16 + (rowQ + xb + n) * C + hi * 16);
#pragma unroll
        for (int t = 0; t < 16; ++t)
            qf[xs][t] = (_Float16)((float)qf[xs][t] * 0.17677669529663687f);
#pragma unroll
        for (int kc = 0; kc < 4; ++kc)
            xnf[xs][kc] = *(const v16h*)(xn + ((size_t)(xb + n) * R + i) * CZ + kc * 32 + hi * 16);
    }

    v8f oacc[2][2];
#pragma unroll
    for (int a = 0; a < 2; ++a)
#pragma unroll
        for (int c = 0; c < 2; ++c)
#pragma unroll
            for (int r = 0; r < 8; ++r) oacc[a][c][r] = 0.0f;
    float m[2] = {-1e30f, -1e30f};
    float l[2] = {0.0f, 0.0f};

#if HAS_TDM
    // prologue: stream slab 0 into buffer 0
    if (wave == 0) {
        tdm_issue(k16 + (rowQ + 0) * C, (unsigned)(uintptr_t)&smK[0][0], 32, 32, 32);
        tdm_issue(vT16 + (size_t)(h * C) * RR + (size_t)i * R, (unsigned)(uintptr_t)&smV[0][0], 32, 32, RR);
        tdm_issue(Wb16 + (size_t)h * R * CZ, (unsigned)(uintptr_t)&smWb[0][0], 128, 32, 128);
    }
#endif

    for (int y = 0; y < R; y += 32) {
#if HAS_TDM
        const int buf = (y >> 5) & 1;
        if (wave == 0) {
            if (y + 32 < R) {   // stream next slab into the other buffer
                const int nb = buf ^ 1;
                const int yn = y + 32;
                tdm_issue(k16 + (rowQ + yn) * C, (unsigned)(uintptr_t)&smK[nb][0], 32, 32, 32);
                tdm_issue(vT16 + (size_t)(h * C) * RR + (size_t)i * R + yn,
                          (unsigned)(uintptr_t)&smV[nb][0], 32, 32, RR);
                tdm_issue(Wb16 + ((size_t)h * R + yn) * CZ, (unsigned)(uintptr_t)&smWb[nb][0], 128, 32, 128);
                __builtin_amdgcn_s_wait_tensorcnt(3);   // this slab's 3 loads done
            } else {
                __builtin_amdgcn_s_wait_tensorcnt(0);
            }
        }
        __syncthreads();                                 // LDS slab ready for all waves
#else
        const int buf = 0;
        __syncthreads();                                 // protect buffer overwrite
        {   // cooperative fill: K (1024 halves), Wb (4096 halves) contiguous; V strided
            const int t = threadIdx.x;
            {   const int idx = t * 4;                   // 8B per thread
                *(unsigned long long*)(&smK[0][0] + idx) =
                    *(const unsigned long long*)(k16 + (rowQ + y) * C + idx);
            }
            {   const int row = t >> 3, c = (t & 7) * 4; // 32 rows x 8 chunks of 8B
                *(unsigned long long*)(&smV[0][0] + row * 32 + c) =
                    *(const unsigned long long*)(vT16 + (size_t)(h * C + row) * RR + (size_t)i * R + y + c);
            }
            {   const int idx = t * 16;                  // 32B per thread
                *(v16h*)(&smWb[0][0] + idx) =
                    *(const v16h*)(Wb16 + ((size_t)h * R + y) * CZ + idx);
            }
        }
        __syncthreads();
#endif
        const _Float16* bK  = &smK[buf][0];
        const _Float16* bV  = &smV[buf][0];
        const _Float16* bWb = &smWb[buf][0];

        // S^T = K * Q^T (shared K fragments from LDS, K dim = C = 32)
        v16h kf0 = load_a_frag(bK, 32);
        v16h kf1 = load_a_frag(bK + 16 * 32, 32);
        v8f s[2][2];
#pragma unroll
        for (int xs = 0; xs < 2; ++xs) {
            s[xs][0] = wmma_f32_f16(kf0, qf[xs], v8f{});
            s[xs][1] = wmma_f32_f16(kf1, qf[xs], v8f{});
        }
        // + pair bias: Wb'(y,z) * xn^T(z,x); Wb' fragments from LDS
#pragma unroll
        for (int kc = 0; kc < 4; ++kc) {
            v16h wb0 = load_a_frag(bWb + kc * 32, CZ);
            v16h wb1 = load_a_frag(bWb + 16 * CZ + kc * 32, CZ);
#pragma unroll
            for (int xs = 0; xs < 2; ++xs) {
                s[xs][0] = wmma_f32_f16(wb0, xnf[xs][kc], s[xs][0]);
                s[xs][1] = wmma_f32_f16(wb1, xnf[xs][kc], s[xs][1]);
            }
        }
        const v8f bb0 = *(const v8f*)(bbf + h * R + y + mb);
        const v8f bb1 = *(const v8f*)(bbf + h * R + y + 16 + mb);
        v16h vf0 = load_a_frag(bV, 32);
        v16h vf1 = load_a_frag(bV + 16 * 32, 32);

        // prefetch slab y+64 toward L2, ahead of the TDM/fill stream
        if (y + 64 < R) {
            __builtin_prefetch(k16 + (rowQ + y + 64 + lane) * C, 0, 1);
            __builtin_prefetch(Wb16 + ((size_t)h * R + y + 64 + lane) * CZ, 0, 1);
            __builtin_prefetch(vT16 + (size_t)(h * C + lane) * RR + (size_t)i * R + y + 64, 0, 1);
        }

#pragma unroll
        for (int xs = 0; xs < 2; ++xs) {
#pragma unroll
            for (int r = 0; r < 8; ++r) { s[xs][0][r] += bb0[r]; s[xs][1][r] += bb1[r]; }
            // online softmax over this 32-wide y slab (per x = column n)
            float tmax = -1e30f;
#pragma unroll
            for (int r = 0; r < 8; ++r) tmax = fmaxf(tmax, fmaxf(s[xs][0][r], s[xs][1][r]));
            tmax = fmaxf(tmax, __shfl_xor(tmax, 16, 32));
            const float mnew  = fmaxf(m[xs], tmax);
            const float alpha = __expf(m[xs] - mnew);
            m[xs] = mnew;
            float p0[8], p1[8], rs = 0.0f;
#pragma unroll
            for (int r = 0; r < 8; ++r) {
                p0[r] = __expf(s[xs][0][r] - mnew);
                p1[r] = __expf(s[xs][1][r] - mnew);
                rs += p0[r] + p1[r];
            }
            rs += __shfl_xor(rs, 16, 32);
            l[xs] = l[xs] * alpha + rs;
            // P^T -> B fragment (K=y in 0..31, N=x); lane<->lane^16 exchange only
            v16h pb;
#pragma unroll
            for (int r = 0; r < 8; ++r) {
                const float o0 = __shfl_xor(p0[r], 16, 32);
                const float o1 = __shfl_xor(p1[r], 16, 32);
                pb[r]     = (_Float16)(hi == 0 ? p0[r] : o1);
                pb[r + 8] = (_Float16)(hi == 0 ? o0 : p1[r]);
            }
            // O^T = O^T*alpha + V^T(c,y) * P^T(y,x)
#pragma unroll
            for (int r = 0; r < 8; ++r) { oacc[xs][0][r] *= alpha; oacc[xs][1][r] *= alpha; }
            oacc[xs][0] = wmma_f32_f16(vf0, pb, oacc[xs][0]);
            oacc[xs][1] = wmma_f32_f16(vf1, pb, oacc[xs][1]);
        }
        __syncthreads();   // all waves done with this buffer before it is restreamed
    }

    // epilogue: normalize, gate, store concat layout ocat[(i*R+x)*128 + h*32 + c]
#pragma unroll
    for (int xs = 0; xs < 2; ++xs) {
        const int xb = xBase + xs * 16;
        const float linv = 1.0f / l[xs];
        const size_t orow = (size_t)i * R + xb + n;
#pragma unroll
        for (int cb = 0; cb < 2; ++cb) {
            v8h gv = *(const v8h*)(g16 + (rowQ + xb + n) * C + cb * 16 + mb);
            v8h ov;
#pragma unroll
            for (int r = 0; r < 8; ++r)
                ov[r] = (_Float16)(oacc[xs][cb][r] * linv * (float)gv[r]);
            *(v8h*)(ocat + orow * (H * C) + h * C + cb * 16 + mb) = ov;
        }
    }
}

// ---------------------------------------------------------------------------
// Kernel 5: output projection, TRANSPOSED: out^T[z][ij] = Wout(z,d)*ocat^T(d,ij)
// + bout. Lane's 8 C-values are contiguous z of one output row -> 32B stores.
// Wout A-fragments reused across 4 ij-tiles. grid (RR/64, 2), block 128.
// ---------------------------------------------------------------------------
__global__ void ta_outproj_kernel(
    const _Float16* __restrict__ ocat, const _Float16* __restrict__ Wout16,
    const float* __restrict__ bout, float* __restrict__ out)
{
    const int wave = threadIdx.x >> 5;
    const int lane = threadIdx.x & 31;
    const int n  = lane & 15;
    const int mb = (lane >> 4) * 8;
    const int zBase = (blockIdx.y * 4 + wave) * 16;

    v16h aw[4];
#pragma unroll
    for (int kc = 0; kc < 4; ++kc)
        aw[kc] = load_a_frag(Wout16 + (size_t)zBase * CZ + kc * 32, CZ);
    const v8f bz = *(const v8f*)(bout + zBase + mb);

    for (int t = 0; t < 4; ++t) {
        const size_t ijBase = (size_t)blockIdx.x * 64 + t * 16;
        v8f acc = bz;
#pragma unroll
        for (int kc = 0; kc < 4; ++kc) {
            v16h bo = load_b_frag(ocat + ijBase * CZ + kc * 32, CZ);
            acc = wmma_f32_f16(aw[kc], bo, acc);
        }
        *(v8f*)(out + (ijBase + n) * CZ + zBase + mb) = acc;   // 32B contiguous
    }
}

// ---------------------------------------------------------------------------
extern "C" void kernel_launch(void* const* d_in, const int* in_sizes, int n_in,
                              void* d_out, int out_size, void* d_ws, size_t ws_size,
                              hipStream_t stream) {
    (void)in_sizes; (void)n_in; (void)out_size; (void)ws_size;
    const float* pair  = (const float*)d_in[0];
    const float* gamma = (const float*)d_in[1];
    const float* beta  = (const float*)d_in[2];
    const float* Wq    = (const float*)d_in[3];
    const float* bq    = (const float*)d_in[4];
    const float* Wk    = (const float*)d_in[5];
    const float* bk    = (const float*)d_in[6];
    const float* Wv    = (const float*)d_in[7];
    const float* bv    = (const float*)d_in[8];
    const float* Wb    = (const float*)d_in[9];
    const float* Wg    = (const float*)d_in[10];
    const float* bg    = (const float*)d_in[11];
    const float* Wout  = (const float*)d_in[12];
    const float* bout  = (const float*)d_in[13];
    float* out = (float*)d_out;

    char* ws = (char*)d_ws;
    size_t off = 0;
    auto alloc = [&](size_t bytes) -> void* {
        void* p = ws + off;
        off += (bytes + 255) & ~(size_t)255;
        return p;
    };
    _Float16* xn16   = (_Float16*)alloc((size_t)RR * CZ * 2);       // 16.8 MB
    _Float16* q16    = (_Float16*)alloc((size_t)H * RR * C * 2);    // 16.8 MB
    _Float16* k16    = (_Float16*)alloc((size_t)H * RR * C * 2);    // 16.8 MB
    _Float16* vT16   = (_Float16*)alloc((size_t)H * C * RR * 2);    // 16.8 MB
    _Float16* g16    = (_Float16*)alloc((size_t)H * RR * C * 2);    // 16.8 MB
    _Float16* ocat16 = (_Float16*)alloc((size_t)RR * CZ * 2);       // 16.8 MB
    _Float16* Wq16   = (_Float16*)alloc((size_t)H * C * CZ * 2);
    _Float16* Wk16   = (_Float16*)alloc((size_t)H * C * CZ * 2);
    _Float16* Wv16   = (_Float16*)alloc((size_t)H * C * CZ * 2);
    _Float16* Wg16   = (_Float16*)alloc((size_t)H * C * CZ * 2);
    _Float16* Wb16   = (_Float16*)alloc((size_t)H * R * CZ * 2);
    _Float16* Wout16 = (_Float16*)alloc((size_t)CZ * (H * C) * 2);
    float* bqf = (float*)alloc((size_t)H * C * 4);
    float* bkf = (float*)alloc((size_t)H * C * 4);
    float* bvf = (float*)alloc((size_t)H * C * 4);
    float* bgf = (float*)alloc((size_t)H * C * 4);
    float* bbf = (float*)alloc((size_t)H * R * 4);

    ta_prep_kernel<<<1664, 128, 0, stream>>>(
        gamma, beta, Wq, bq, Wk, bk, Wv, bv, Wb, Wg, bg, Wout,
        Wq16, Wk16, Wv16, Wg16, Wb16, Wout16, bqf, bkf, bvf, bgf, bbf);

    ta_ln_kernel<<<RR, 128, 0, stream>>>(pair, xn16);

    ta_proj_kernel<<<dim3(RR / 64, H), 128, 0, stream>>>(
        xn16, Wq16, Wk16, Wv16, Wg16, bqf, bkf, bvf, bgf, q16, k16, vT16, g16);

    ta_attn_kernel<<<dim3(R, H), 256, 0, stream>>>(
        xn16, q16, k16, vT16, g16, Wb16, bbf, ocat16);

    ta_outproj_kernel<<<dim3(RR / 64, 2), 128, 0, stream>>>(ocat16, Wout16, bout, out);
}
